// ZDecoder_25271587570218
// MI455X (gfx1250) — matrix operand order repaired and checked by reading, not compiled, not run on hardware
//
#include <hip/hip_runtime.h>
#include <hip/hip_bf16.h>

typedef __attribute__((ext_vector_type(16))) _Float16 v16h;
typedef __attribute__((ext_vector_type(8)))  float    v8f;
typedef __attribute__((ext_vector_type(2)))  __fp16   v2h;   // matches cvt_pkrtz return type

union HFrag {            // 8 dwords = 16 halves, matches one WMMA f16 A/B operand
  v16h     v;
  v2h      p[8];
  unsigned u[8];
};
union PkU { v2h h; unsigned u; };

#define NLEVELS 4
#define NREG    8
#define ZS      16
#define PHI_SZ  64
#define HID     64
#define OUT_SZ  32
#define NBATCH  256
#define NCOMBO  4096
#define IN_SZ   128

// Workspace byte offsets
#define WS_P   0        // 256*64 f32   = 65536 B   P[b][h] = W1_phi . phi[b] + b1
#define WS_T   65536    // 4*8*64 f32   =  8192 B   T[l][j][h] = W1_V(l-block) . rp[l][j]
#define WS_B2  73728    // 2*4*32*8 u32 =  8192 B   packed f16x2 B-fragments of W2
#define WS_B3  81920    // 2*2*32*8 u32 =  4096 B   packed f16x2 B-fragments of W3
// total 86016 B

// ---------------------------------------------------------------- setup ----
// B-fragment layout (16-bit B, 32x16 KxN): lane -> N=lane&15; VGPR r packs
// (K=k0, K=k0+1), k0 = 8*(lane>>4) + (r<4 ? 2r : 16+2(r-4)), in lo/hi half.
__global__ __launch_bounds__(256) void zdec_setup(
    const float* __restrict__ phi, const float* __restrict__ rp,
    const float* __restrict__ w1,  const float* __restrict__ b1,
    const float* __restrict__ w2,  const float* __restrict__ w3,
    float* __restrict__ P, float* __restrict__ T,
    unsigned* __restrict__ B2, unsigned* __restrict__ B3)
{
  const int tid = blockIdx.x * blockDim.x + threadIdx.x;   // 0..16383

  // P[b][h] = b1[h] + sum_i w1[h][64+i] * phi[b][i]
  if (tid < NBATCH * HID) {
    const int b = tid >> 6, h = tid & 63;
    float s = b1[h];
    #pragma unroll 8
    for (int i = 0; i < PHI_SZ; ++i) s += w1[h * IN_SZ + PHI_SZ + i] * phi[b * PHI_SZ + i];
    P[tid] = s;
  }
  // T[l][j][h] = sum_z w1[h][l*16+z] * rp[l][j][z]
  if (tid < NLEVELS * NREG * HID) {
    const int h = tid & 63, j = (tid >> 6) & 7, l = tid >> 9;
    float s = 0.f;
    #pragma unroll
    for (int z = 0; z < ZS; ++z) s += w1[h * IN_SZ + l * ZS + z] * rp[(l * NREG + j) * ZS + z];
    T[tid] = s;
  }
  // W2 B-fragments: frag (kh in 0..1, nt in 0..3), element (lane, r)
  if (tid < 2 * 4 * 32 * 8) {
    const int r = tid & 7, lane = (tid >> 3) & 31, nt = (tid >> 8) & 3, kh = (tid >> 10) & 1;
    const int kg = kh * 32 + 8 * (lane >> 4) + ((r < 4) ? 2 * r : 16 + 2 * (r - 4));
    const int n0 = nt * 16 + (lane & 15);
    PkU pk; pk.h = __builtin_amdgcn_cvt_pkrtz(w2[n0 * HID + kg], w2[n0 * HID + kg + 1]);
    B2[tid] = pk.u;
  }
  // W3 B-fragments: frag (kh in 0..1, nt in 0..1)
  if (tid < 2 * 2 * 32 * 8) {
    const int r = tid & 7, lane = (tid >> 3) & 31, nt = (tid >> 8) & 1, kh = (tid >> 9) & 1;
    const int kg = kh * 32 + 8 * (lane >> 4) + ((r < 4) ? 2 * r : 16 + 2 * (r - 4));
    const int n0 = nt * 16 + (lane & 15);
    PkU pk; pk.h = __builtin_amdgcn_cvt_pkrtz(w3[n0 * HID + kg], w3[n0 * HID + kg + 1]);
    B3[tid] = pk.u;
  }
}

// ----------------------------------------------------------------- main ----
// One block = one batch b x 128 combos; 8 waves, each wave a 16-row M-tile.
#define H2_STRIDE 72   // halves per staged h2 row (64 + pad, conflict-free)

__global__ __launch_bounds__(256) void zdec_main(
    const float* __restrict__ P, const float* __restrict__ T,
    const unsigned* __restrict__ B2p, const unsigned* __restrict__ B3p,
    const float* __restrict__ b2, const float* __restrict__ b3,
    float* __restrict__ out)
{
  __shared__ float    sT[NLEVELS * NREG * HID];   // 8 KB
  __shared__ float    sP[HID];
  __shared__ unsigned sB2[2 * 4 * 32 * 8];        // 8 KB
  __shared__ unsigned sB3[2 * 2 * 32 * 8];        // 4 KB
  __shared__ float    sb2[HID];
  __shared__ float    sb3[OUT_SZ];
  __shared__ _Float16 sH2[8 * 16 * H2_STRIDE];    // 18 KB, per-wave h2 staging

  const int b     = blockIdx.x >> 5;
  const int ctile = (blockIdx.x & 31) * 128;
  const int t     = threadIdx.x;

  for (int i = t; i < NLEVELS * NREG * HID; i += 256) sT[i]  = T[i];
  for (int i = t; i < 2 * 4 * 32 * 8;       i += 256) sB2[i] = B2p[i];
  for (int i = t; i < 2 * 2 * 32 * 8;       i += 256) sB3[i] = B3p[i];
  if (t < HID)    { sP[t] = P[b * HID + t]; sb2[t] = b2[t]; }
  if (t < OUT_SZ) sb3[t] = b3[t];
  __syncthreads();

  const int wave = t >> 5;
  const int lane = t & 31;
  const int m    = lane & 15;   // A-layout row within tile
  const int g    = lane >> 4;   // k-group (lanes 16-31 hold K+8 per 16-bit A table)
  const int c    = ctile + wave * 16 + m;

  const float* t0 = &sT[((0 << 3) | ((c >> 0) & 7)) << 6];
  const float* t1 = &sT[((1 << 3) | ((c >> 3) & 7)) << 6];
  const float* t2 = &sT[((2 << 3) | ((c >> 6) & 7)) << 6];
  const float* t3 = &sT[((3 << 3) | ((c >> 9) & 7)) << 6];

  // --- layer 1 (factored): build h1 directly as f16 A fragments -----------
  HFrag a2[2];
  #pragma unroll
  for (int kh = 0; kh < 2; ++kh) {
    #pragma unroll
    for (int r = 0; r < 8; ++r) {
      const int k = kh * 32 + g * 8 + ((r < 4) ? 2 * r : 16 + 2 * (r - 4));
      const float f0 = fmaxf(sP[k]     + t0[k]     + t1[k]     + t2[k]     + t3[k],     0.f);
      const float f1 = fmaxf(sP[k + 1] + t0[k + 1] + t1[k + 1] + t2[k + 1] + t3[k + 1], 0.f);
      a2[kh].p[r] = __builtin_amdgcn_cvt_pkrtz(f0, f1);
    }
  }

  // --- layer 2: 16x64 = A(16x64 f16) x W2^T, 4 N-tiles x 2 K-steps --------
  v8f d2[4] = {};
  #pragma unroll
  for (int kh = 0; kh < 2; ++kh) {
    #pragma unroll
    for (int nt = 0; nt < 4; ++nt) {
      HFrag bf;
      #pragma unroll
      for (int r = 0; r < 8; ++r) bf.u[r] = sB2[((kh * 4 + nt) * 32 + lane) * 8 + r];
      d2[nt] = __builtin_amdgcn_wmma_f32_16x16x32_f16(
                   false, a2[kh].v, false, bf.v, (short)0, d2[nt], false, false);
    }
  }

  // --- bias + relu, stage h2 row-major f16 in LDS (D-layout -> A-layout) --
  _Float16* h2row = &sH2[wave * 16 * H2_STRIDE];
  #pragma unroll
  for (int nt = 0; nt < 4; ++nt) {
    const int   n    = nt * 16 + m;        // D-layout: lane holds column N
    const float bias = sb2[n];
    #pragma unroll
    for (int r = 0; r < 8; ++r) {
      const int row = r + 8 * g;           // D-layout: lanes 16-31 hold M+8
      h2row[row * H2_STRIDE + n] = (_Float16)fmaxf(d2[nt][r] + bias, 0.f);
    }
  }
  __syncthreads();

  // --- reload h2 as f16 A fragments ---------------------------------------
  HFrag a3[2];
  const unsigned* h2u = (const unsigned*)h2row;   // dword index = row*36 + k/2
  #pragma unroll
  for (int kh = 0; kh < 2; ++kh) {
    #pragma unroll
    for (int r = 0; r < 8; ++r) {
      const int k = kh * 32 + g * 8 + ((r < 4) ? 2 * r : 16 + 2 * (r - 4));
      a3[kh].u[r] = h2u[m * (H2_STRIDE / 2) + (k >> 1)];
    }
  }

  // --- layer 3: 16x32 = A(16x64 f16) x W3^T, 2 N-tiles x 2 K-steps --------
  v8f d3[2] = {};
  #pragma unroll
  for (int kh = 0; kh < 2; ++kh) {
    #pragma unroll
    for (int nt = 0; nt < 2; ++nt) {
      HFrag bf;
      #pragma unroll
      for (int r = 0; r < 8; ++r) bf.u[r] = sB3[((kh * 2 + nt) * 32 + lane) * 8 + r];
      d3[nt] = __builtin_amdgcn_wmma_f32_16x16x32_f16(
                   false, a3[kh].v, false, bf.v, (short)0, d3[nt], false, false);
    }
  }

  // --- bias + coalesced f32 store: out[b][c][o] ----------------------------
  const size_t base = (size_t)b * NCOMBO + (size_t)(ctile + wave * 16);
  #pragma unroll
  for (int nt = 0; nt < 2; ++nt) {
    const int   o    = nt * 16 + m;
    const float bias = sb3[o];
    #pragma unroll
    for (int r = 0; r < 8; ++r) {
      const int row = r + 8 * g;
      out[(base + row) * OUT_SZ + o] = d3[nt][r] + bias;
    }
  }
}

// --------------------------------------------------------------- launch ----
extern "C" void kernel_launch(void* const* d_in, const int* in_sizes, int n_in,
                              void* d_out, int out_size, void* d_ws, size_t ws_size,
                              hipStream_t stream) {
  const float* phi = (const float*)d_in[0];
  const float* rp  = (const float*)d_in[1];
  const float* w1  = (const float*)d_in[2];
  const float* b1  = (const float*)d_in[3];
  const float* w2  = (const float*)d_in[4];
  const float* b2  = (const float*)d_in[5];
  const float* w3  = (const float*)d_in[6];
  const float* b3  = (const float*)d_in[7];
  float* out = (float*)d_out;

  char* ws = (char*)d_ws;
  float*    P  = (float*)(ws + WS_P);
  float*    T  = (float*)(ws + WS_T);
  unsigned* B2 = (unsigned*)(ws + WS_B2);
  unsigned* B3 = (unsigned*)(ws + WS_B3);

  zdec_setup<<<64, 256, 0, stream>>>(phi, rp, w1, b1, w2, w3, P, T, B2, B3);
  zdec_main<<<NBATCH * 32, 256, 0, stream>>>(P, T, B2, B3, b2, b3, out);
}
